// TSAE_46102178955328
// MI455X (gfx1250) — matrix operand order repaired and checked by reading, not compiled, not run on hardware
//
#include <hip/hip_runtime.h>
#include <hip/hip_bf16.h>

// ---------------------------------------------------------------------------
// Types / helpers
// ---------------------------------------------------------------------------
typedef float  v8f   __attribute__((ext_vector_type(8)));
typedef __bf16 v16bf __attribute__((ext_vector_type(16)));
typedef __bf16 v8bf  __attribute__((ext_vector_type(8)));

union Frag16 { v16bf v; v8bf h[2]; };

__device__ inline unsigned short f2bfu(float x) {
    union { float f; unsigned u; } c; c.f = x;
    unsigned r = c.u + 0x7FFFu + ((c.u >> 16) & 1u);
    return (unsigned short)(r >> 16);
}
__device__ inline float bfu2f(unsigned short s) {
    union { unsigned u; float f; } c; c.u = ((unsigned)s) << 16;
    return c.f;
}
__device__ inline v8f v8f_zero() {
    v8f r;
#pragma unroll
    for (int i = 0; i < 8; ++i) r[i] = 0.0f;
    return r;
}

// CDNA5 async global->LDS staging (ASYNCcnt-tracked DMA, no VGPR round trip)
__device__ inline unsigned lds_off32(const void* p) {
    return (unsigned)(unsigned long long)(__attribute__((address_space(3))) const char*)p;
}
__device__ inline void async_ld_b128(const void* lds, const void* g) {
    asm volatile("global_load_async_to_lds_b128 %0, %1, off"
                 :: "v"(lds_off32(lds)), "v"((unsigned long long)g)
                 : "memory");
}
__device__ inline void wait_async0() {
    asm volatile("s_wait_asynccnt 0x0" ::: "memory");
}

#define Bb   4
#define Dd   8
#define Ls   512
#define Hh   768
#define Ff   4096
#define NH   8
#define HD   96
#define M1   16384      /* B*D*L rows for stage GEMMs */
#define MR   14336      /* B*(D-1)*L residual rows */
#define NPRED 11010048  /* MR*Hh */
#define NZ    58720256  /* MR*Ff */

// ---------------------------------------------------------------------------
// f32 -> bf16 converter
// ---------------------------------------------------------------------------
__global__ void conv_bf16(const float* __restrict__ in, unsigned short* __restrict__ out, int n) {
    int i = blockIdx.x * 256 + threadIdx.x;
    if (i < n) out[i] = f2bfu(in[i]);
}

// ---------------------------------------------------------------------------
// LayerNorm (per row of H=768), bf16 output
// ---------------------------------------------------------------------------
__launch_bounds__(256)
__global__ void layernorm_bf16(const float* __restrict__ x, const float* __restrict__ g,
                               const float* __restrict__ b, unsigned short* __restrict__ out) {
    __shared__ float red[256];
    const int row = blockIdx.x;
    const int tid = threadIdx.x;
    const float* xr = x + (size_t)row * Hh;

    float s = 0.f;
    for (int i = tid; i < Hh; i += 256) s += xr[i];
    red[tid] = s; __syncthreads();
    for (int st = 128; st > 0; st >>= 1) { if (tid < st) red[tid] += red[tid + st]; __syncthreads(); }
    float mu = red[0] / (float)Hh;
    __syncthreads();

    float s2 = 0.f;
    for (int i = tid; i < Hh; i += 256) { float d = xr[i] - mu; s2 += d * d; }
    red[tid] = s2; __syncthreads();
    for (int st = 128; st > 0; st >>= 1) { if (tid < st) red[tid] += red[tid + st]; __syncthreads(); }
    float rstd = rsqrtf(red[0] / (float)Hh + 1e-5f);

    for (int i = tid; i < Hh; i += 256)
        out[(size_t)row * Hh + i] = f2bfu((xr[i] - mu) * rstd * g[i] + b[i]);
}

// ---------------------------------------------------------------------------
// Tiled bf16 WMMA GEMM: C[M,N] = A[M,K] * B + (bias) + (resid), opt relu
//   transB=1: B[k][n] = Bw[n*K + k]   (i.e. X @ W^T with row-major W[N,K])
//   transB=0: B[k][n] = Bw[k*N + n]   (dictionary encode)
// Workgroup tile 128x64, 8 waves in a 4x2 grid; each wave computes 32x32 via
// 2 A-frags x 2 B-frags -> 4 WMMAs (operand reuse doubles math per LDS byte).
// A (and B when transB) staged with global_load_async_to_lds_b128.
// ---------------------------------------------------------------------------
__launch_bounds__(256)
__global__ void gemm_bf16_wmma(const unsigned short* __restrict__ A,
                               const unsigned short* __restrict__ Bw,
                               int M, int N, int K, int transB,
                               const float* __restrict__ bias, int relu,
                               const float* __restrict__ resid,
                               float* __restrict__ outF,
                               unsigned short* __restrict__ outH) {
    __shared__ unsigned short As[128][32];
    __shared__ unsigned short Bs[64][32];   // stored [n][k] for contiguous fragments
    const int tid  = threadIdx.x;
    const int lane = tid & 31, wid = tid >> 5;
    const int m0 = blockIdx.y * 128, n0 = blockIdx.x * 64;
    const int mo = (wid & 3) * 32, no = (wid >> 2) * 32;
    const int r  = lane & 15, hf = lane >> 4;

    v8f a00 = v8f_zero(), a01 = v8f_zero();
    v8f a10 = v8f_zero(), a11 = v8f_zero();

    const int arow = tid >> 1, akc = (tid & 1) * 16;   // A: 128 rows x 32 k, 16 elems/thread
    const int brow = tid >> 2, bkc = (tid & 3) * 8;    // B(transB): 64 rows x 32 k, 8 elems/thread

    for (int k0 = 0; k0 < K; k0 += 32) {
        __syncthreads();
        async_ld_b128(&As[arow][akc],     A + (size_t)(m0 + arow) * K + k0 + akc);
        async_ld_b128(&As[arow][akc + 8], A + (size_t)(m0 + arow) * K + k0 + akc + 8);
        if (transB) {
            async_ld_b128(&Bs[brow][bkc], Bw + (size_t)(n0 + brow) * K + k0 + bkc);
        } else {
            int kk = tid >> 3, nc = (tid & 7) * 8;
            uint4 d = *(const uint4*)(Bw + (size_t)(k0 + kk) * N + n0 + nc);
            const unsigned short* ds = (const unsigned short*)&d;
#pragma unroll
            for (int e = 0; e < 8; ++e) Bs[nc + e][kk] = ds[e];
        }
        wait_async0();
        __syncthreads();

        if (k0 + 32 < K)   // hint next A tile -> global_prefetch
            __builtin_prefetch(A + (size_t)(m0 + arow) * K + k0 + 32, 0, 1);

        Frag16 af0, af1, bf0, bf1;
        af0.h[0] = *(const v8bf*)&As[mo + r][hf * 8];
        af0.h[1] = *(const v8bf*)&As[mo + r][16 + hf * 8];
        af1.h[0] = *(const v8bf*)&As[mo + 16 + r][hf * 8];
        af1.h[1] = *(const v8bf*)&As[mo + 16 + r][16 + hf * 8];
        bf0.h[0] = *(const v8bf*)&Bs[no + r][hf * 8];
        bf0.h[1] = *(const v8bf*)&Bs[no + r][16 + hf * 8];
        bf1.h[0] = *(const v8bf*)&Bs[no + 16 + r][hf * 8];
        bf1.h[1] = *(const v8bf*)&Bs[no + 16 + r][16 + hf * 8];

        a00 = __builtin_amdgcn_wmma_f32_16x16x32_bf16(false, af0.v, false, bf0.v, (short)0, a00, false, false);
        a01 = __builtin_amdgcn_wmma_f32_16x16x32_bf16(false, af0.v, false, bf1.v, (short)0, a01, false, false);
        a10 = __builtin_amdgcn_wmma_f32_16x16x32_bf16(false, af1.v, false, bf0.v, (short)0, a10, false, false);
        a11 = __builtin_amdgcn_wmma_f32_16x16x32_bf16(false, af1.v, false, bf1.v, (short)0, a11, false, false);
    }

#pragma unroll
    for (int i = 0; i < 8; ++i) {
#pragma unroll
        for (int mt = 0; mt < 2; ++mt) {
            int row = m0 + mo + mt * 16 + i + hf * 8;
#pragma unroll
            for (int nt = 0; nt < 2; ++nt) {
                int col = n0 + no + nt * 16 + r;
                float vv;
                if (mt == 0) vv = (nt == 0) ? a00[i] : a01[i];
                else         vv = (nt == 0) ? a10[i] : a11[i];
                if (bias)  vv += bias[col];
                if (relu)  vv = fmaxf(vv, 0.0f);
                if (resid) vv += resid[(size_t)row * N + col];
                if (outF)  outF[(size_t)row * N + col] = vv;
                else       outH[(size_t)row * N + col] = f2bfu(vv);
            }
        }
    }
}

// ---------------------------------------------------------------------------
// Flash attention over L=512 (non-causal), head dim 96, bf16 q/k/v.
// One workgroup handles one (seq,head) * 128 query rows; each wave = 16 rows.
// K/V tiles (32 keys) staged cooperatively in LDS; online softmax; P.V by WMMA.
// ---------------------------------------------------------------------------
__launch_bounds__(256)
__global__ void attn_seq(const unsigned short* __restrict__ q,
                         const unsigned short* __restrict__ k,
                         const unsigned short* __restrict__ v,
                         unsigned short* __restrict__ out) {
    __shared__ unsigned short Ks[32][96];        // [key][d]
    __shared__ unsigned short Vs[96][32];        // [d][key]  (transposed for B frags)
    __shared__ unsigned short Ps[8][16][32];     // per-wave P staging
    const int tid = threadIdx.x, lane = tid & 31, wid = tid >> 5;
    const int sh = blockIdx.x >> 2;
    const int s = sh >> 3, h = sh & 7;
    const int q0 = (blockIdx.x & 3) * 128 + wid * 16;
    const int r = lane & 15, hf = lane >> 4;
    const float scale = 0.10206207f;  // 1/sqrt(96)

    Frag16 qf[3];
#pragma unroll
    for (int c = 0; c < 3; ++c) {
        const unsigned short* p = q + (size_t)(s * Ls + q0 + r) * Hh + h * HD + c * 32 + hf * 8;
        qf[c].h[0] = *(const v8bf*)p;
        qf[c].h[1] = *(const v8bf*)(p + 16);
    }

    float m[8], l[8];
    v8f o[6];
#pragma unroll
    for (int i = 0; i < 8; ++i) { m[i] = -1e30f; l[i] = 0.f; }
#pragma unroll
    for (int t = 0; t < 6; ++t) o[t] = v8f_zero();

    for (int j0 = 0; j0 < Ls; j0 += 32) {
        __syncthreads();
        {   // cooperative K/V staging: 32 keys x 96 dims
            int j = tid >> 3, base = (tid & 7) * 12;
            const unsigned short* kp = k + (size_t)(s * Ls + j0 + j) * Hh + h * HD + base;
            const unsigned short* vp = v + (size_t)(s * Ls + j0 + j) * Hh + h * HD + base;
#pragma unroll
            for (int e = 0; e < 12; ++e) { Ks[j][base + e] = kp[e]; Vs[base + e][j] = vp[e]; }
        }
        __syncthreads();

        v8f S0 = v8f_zero(), S1 = v8f_zero();
#pragma unroll
        for (int c = 0; c < 3; ++c) {
            Frag16 b0, b1;
            b0.h[0] = *(const v8bf*)&Ks[r][c * 32 + hf * 8];
            b0.h[1] = *(const v8bf*)&Ks[r][c * 32 + 16 + hf * 8];
            b1.h[0] = *(const v8bf*)&Ks[16 + r][c * 32 + hf * 8];
            b1.h[1] = *(const v8bf*)&Ks[16 + r][c * 32 + 16 + hf * 8];
            S0 = __builtin_amdgcn_wmma_f32_16x16x32_bf16(false, qf[c].v, false, b0.v, (short)0, S0, false, false);
            S1 = __builtin_amdgcn_wmma_f32_16x16x32_bf16(false, qf[c].v, false, b1.v, (short)0, S1, false, false);
        }

        float p0[8], p1[8], sc[8];
#pragma unroll
        for (int i = 0; i < 8; ++i) {
            float a  = S0[i] * scale, bb = S1[i] * scale;
            float t  = fmaxf(a, bb);
            t = fmaxf(t, __shfl_xor(t, 1));
            t = fmaxf(t, __shfl_xor(t, 2));
            t = fmaxf(t, __shfl_xor(t, 4));
            t = fmaxf(t, __shfl_xor(t, 8));
            float mn = fmaxf(m[i], t);
            sc[i] = __expf(m[i] - mn);
            m[i]  = mn;
            p0[i] = __expf(a - mn);
            p1[i] = __expf(bb - mn);
            float rs = p0[i] + p1[i];
            rs += __shfl_xor(rs, 1); rs += __shfl_xor(rs, 2);
            rs += __shfl_xor(rs, 4); rs += __shfl_xor(rs, 8);
            l[i] = l[i] * sc[i] + rs;
        }
#pragma unroll
        for (int t = 0; t < 6; ++t)
#pragma unroll
            for (int i = 0; i < 8; ++i) o[t][i] *= sc[i];

        // stage P (f32 C-layout -> bf16 A-layout via LDS)
#pragma unroll
        for (int i = 0; i < 8; ++i) {
            Ps[wid][i + hf * 8][r]      = f2bfu(p0[i]);
            Ps[wid][i + hf * 8][16 + r] = f2bfu(p1[i]);
        }
        __syncthreads();

        Frag16 pa;
        pa.h[0] = *(const v8bf*)&Ps[wid][r][hf * 8];
        pa.h[1] = *(const v8bf*)&Ps[wid][r][16 + hf * 8];
#pragma unroll
        for (int t = 0; t < 6; ++t) {
            Frag16 vb;
            vb.h[0] = *(const v8bf*)&Vs[t * 16 + r][hf * 8];
            vb.h[1] = *(const v8bf*)&Vs[t * 16 + r][16 + hf * 8];
            o[t] = __builtin_amdgcn_wmma_f32_16x16x32_bf16(false, pa.v, false, vb.v, (short)0, o[t], false, false);
        }
    }

#pragma unroll
    for (int i = 0; i < 8; ++i) {
        float inv = 1.0f / l[i];
        int row = s * Ls + q0 + i + hf * 8;
#pragma unroll
        for (int t = 0; t < 6; ++t)
            out[(size_t)row * Hh + h * HD + t * 16 + r] = f2bfu(o[t][i] * inv);
    }
}

// ---------------------------------------------------------------------------
// Tiny causal attention over D=8 (one thread per (row,head,t))
// ---------------------------------------------------------------------------
__launch_bounds__(256)
__global__ void attn_depth(const unsigned short* __restrict__ q,
                           const unsigned short* __restrict__ k,
                           const unsigned short* __restrict__ v,
                           unsigned short* __restrict__ out) {
    int u = blockIdx.x * 256 + threadIdx.x;     // 2048 * 8 * 8 units
    int bl = u >> 6; int rest = u & 63; int h = rest >> 3; int t = rest & 7;
    const float scale = 0.10206207f;
    const unsigned short* qr = q + (size_t)(bl * 8 + t) * Hh + h * HD;

    float sc[8];
    float mx = -1e30f;
#pragma unroll
    for (int j = 0; j < 8; ++j) {
        float d = 0.f;
        if (j <= t) {
            const unsigned short* kr = k + (size_t)(bl * 8 + j) * Hh + h * HD;
            for (int e = 0; e < HD; ++e) d += bfu2f(qr[e]) * bfu2f(kr[e]);
            d *= scale;
            mx = fmaxf(mx, d);
        }
        sc[j] = d;
    }
    float sum = 0.f;
#pragma unroll
    for (int j = 0; j < 8; ++j) {
        sc[j] = (j <= t) ? __expf(sc[j] - mx) : 0.f;
        sum += sc[j];
    }
    float inv = 1.f / sum;
    unsigned short* orow = out + (size_t)(bl * 8 + t) * Hh + h * HD;
    for (int e = 0; e < HD; ++e) {
        float acc = 0.f;
#pragma unroll
        for (int j = 0; j < 8; ++j)
            acc += sc[j] * bfu2f(v[(size_t)(bl * 8 + j) * Hh + h * HD + e]);
        orow[e] = f2bfu(acc * inv);
    }
}

// ---------------------------------------------------------------------------
// (B,D,L,H) <-> (B,L,D,H) transpose; back=0 forward, back=1 reverse
// ---------------------------------------------------------------------------
__global__ void transpose_dl(const float* __restrict__ in, float* __restrict__ out, int back) {
    size_t i = (size_t)blockIdx.x * 256 + threadIdx.x;
    int h = (int)(i % Hh);
    size_t row = i / Hh;
    int b = (int)(row >> 12);
    int d = (int)((row >> 9) & 7);
    int l = (int)(row & 511);
    size_t orow = ((size_t)(b * Ls + l)) * 8 + d;
    if (back) out[i] = in[orow * Hh + h];
    else      out[orow * Hh + h] = in[i];
}

// ---------------------------------------------------------------------------
// residual = x_target - x_pred (bf16 for encode GEMM) + pred-loss partial
// ---------------------------------------------------------------------------
__launch_bounds__(256)
__global__ void residual_pred(const float* __restrict__ x, unsigned short* __restrict__ res,
                              float* __restrict__ acc) {
    __shared__ float red[256];
    size_t i = (size_t)blockIdx.x * 256 + threadIdx.x;     // over MR*Hh
    int h = (int)(i % Hh);
    size_t row = i / Hh;
    int l = (int)(row % Ls);
    size_t bd = row / Ls;
    int dm1 = (int)(bd % 7);
    int b = (int)(bd / 7);
    size_t ip = ((size_t)((b * 8 + dm1)     * Ls + l)) * Hh + h;
    size_t it = ((size_t)((b * 8 + dm1 + 1) * Ls + l)) * Hh + h;
    float dlt = x[it] - x[ip];
    res[i] = f2bfu(dlt);
    red[threadIdx.x] = dlt * dlt;
    __syncthreads();
    for (int st = 128; st > 0; st >>= 1) {
        if (threadIdx.x < st) red[threadIdx.x] += red[threadIdx.x + st];
        __syncthreads();
    }
    if (threadIdx.x == 0) atomicAdd(acc, red[0]);
}

// ---------------------------------------------------------------------------
// In-place top-k (k=64) per row of z (F=4096): binary-search threshold on count
// ---------------------------------------------------------------------------
__launch_bounds__(256)
__global__ void topk_threshold(float* __restrict__ z) {
    __shared__ float zr[Ff];
    __shared__ float fred[256];
    __shared__ int cnt;
    float* row = z + (size_t)blockIdx.x * Ff;
    int tid = threadIdx.x;

    float mx = 0.f;
    for (int i = tid; i < Ff; i += 256) { float vv = row[i]; zr[i] = vv; mx = fmaxf(mx, vv); }
    fred[tid] = mx; __syncthreads();
    for (int st = 128; st > 0; st >>= 1) { if (tid < st) fred[tid] = fmaxf(fred[tid], fred[tid + st]); __syncthreads(); }

    float lo = 0.f, hi = fred[0];
    for (int it = 0; it < 24; ++it) {
        float mid = 0.5f * (lo + hi);
        if (tid == 0) cnt = 0;
        __syncthreads();
        int c = 0;
        for (int i = tid; i < Ff; i += 256) c += (zr[i] > mid) ? 1 : 0;
        atomicAdd(&cnt, c);
        __syncthreads();
        if (cnt > 64) lo = mid; else hi = mid;
        __syncthreads();
    }
    float thr = hi;
    for (int i = tid; i < Ff; i += 256) row[i] = (zr[i] > thr) ? zr[i] : 0.f;
}

// ---------------------------------------------------------------------------
// Sparse decode: x_novel = z @ dict.T via nonzero gather; x_hat; loss partials
// ---------------------------------------------------------------------------
__launch_bounds__(256)
__global__ void decode_sparse(const float* __restrict__ z, const float* __restrict__ dict,
                              const float* __restrict__ x, float* __restrict__ xhat,
                              float* __restrict__ xnovel, float* __restrict__ reconAcc,
                              float* __restrict__ sparseAcc) {
    __shared__ float svals[96];
    __shared__ int   sidx[96];
    __shared__ int   cnt;
    __shared__ float red[256];
    int row = blockIdx.x;
    int tid = threadIdx.x;
    if (tid == 0) cnt = 0;
    __syncthreads();

    const float* zr = z + (size_t)row * Ff;
    float ssum = 0.f;
    for (int i = tid; i < Ff; i += 256) {
        float vv = zr[i];
        if (vv > 0.f) {
            int p = atomicAdd(&cnt, 1);
            if (p < 96) { svals[p] = vv; sidx[p] = i; }
            ssum += vv;
        }
    }
    red[tid] = ssum; __syncthreads();
    for (int st = 128; st > 0; st >>= 1) { if (tid < st) red[tid] += red[tid + st]; __syncthreads(); }
    if (tid == 0) atomicAdd(sparseAcc, red[0]);

    int n = cnt; if (n > 96) n = 96;
    int l = row % Ls;
    int bd = row / Ls;
    int dm1 = bd % 7;
    int b = bd / 7;
    size_t ipbase = ((size_t)((b * 8 + dm1)     * Ls + l)) * Hh;
    size_t itbase = ((size_t)((b * 8 + dm1 + 1) * Ls + l)) * Hh;

    __syncthreads();
    float rsum = 0.f;
    for (int h = tid; h < Hh; h += 256) {
        float acc2 = 0.f;
        for (int j = 0; j < n; ++j) acc2 += svals[j] * dict[(size_t)h * Ff + sidx[j]];
        float pred = x[ipbase + h], tgt = x[itbase + h];
        float xh = pred + acc2;
        xnovel[(size_t)row * Hh + h] = acc2;
        xhat[(size_t)row * Hh + h]   = xh;
        float dd = xh - tgt;
        rsum += dd * dd;
    }
    red[tid] = rsum; __syncthreads();
    for (int st = 128; st > 0; st >>= 1) { if (tid < st) red[tid] += red[tid + st]; __syncthreads(); }
    if (tid == 0) atomicAdd(reconAcc, red[0]);
}

__global__ void zero3(float* a) { if (threadIdx.x < 3) a[threadIdx.x] = 0.f; }

__global__ void final_loss(const float* __restrict__ a, float* __restrict__ out) {
    if (threadIdx.x == 0)
        out[0] = a[0] / (float)NPRED + a[1] / (float)NPRED + 0.001f * (a[2] / (float)NZ);
}

// ---------------------------------------------------------------------------
// Host launcher
// ---------------------------------------------------------------------------
extern "C" void kernel_launch(void* const* d_in, const int* in_sizes, int n_in,
                              void* d_out, int out_size, void* d_ws, size_t ws_size,
                              hipStream_t stream) {
    (void)in_sizes; (void)n_in; (void)out_size; (void)ws_size;

    const float* zL    = (const float*)d_in[0];
    const float* Wq_l  = (const float*)d_in[1];
    const float* Wk_l  = (const float*)d_in[2];
    const float* Wv_l  = (const float*)d_in[3];
    const float* Wo_l  = (const float*)d_in[4];
    const float* g_l   = (const float*)d_in[5];
    const float* b_l   = (const float*)d_in[6];
    const float* Wq_d  = (const float*)d_in[7];
    const float* Wk_d  = (const float*)d_in[8];
    const float* Wv_d  = (const float*)d_in[9];
    const float* Wo_d  = (const float*)d_in[10];
    const float* g_d   = (const float*)d_in[11];
    const float* b_d   = (const float*)d_in[12];
    const float* dict  = (const float*)d_in[13];
    const float* biasn = (const float*)d_in[14];

    // workspace carve-up
    char* w = (char*)d_ws;
    float* accum = (float*)w;                       w += 256;
    float* x_cur = (float*)w;                       w += (size_t)M1 * Hh * 4;
    float* x2    = (float*)w;                       w += (size_t)M1 * Hh * 4;
    unsigned short* lnb   = (unsigned short*)w;     w += (size_t)M1 * Hh * 2;
    unsigned short* qb    = (unsigned short*)w;     w += (size_t)M1 * Hh * 2;
    unsigned short* kb    = (unsigned short*)w;     w += (size_t)M1 * Hh * 2;
    unsigned short* vb    = (unsigned short*)w;     w += (size_t)M1 * Hh * 2;
    unsigned short* attnb = (unsigned short*)w;     w += (size_t)M1 * Hh * 2;
    unsigned short* resb  = (unsigned short*)w;     w += (size_t)MR * Hh * 2;
    unsigned short* wqlb  = (unsigned short*)w;     w += (size_t)Hh * Hh * 2;
    unsigned short* wklb  = (unsigned short*)w;     w += (size_t)Hh * Hh * 2;
    unsigned short* wvlb  = (unsigned short*)w;     w += (size_t)Hh * Hh * 2;
    unsigned short* wolb  = (unsigned short*)w;     w += (size_t)Hh * Hh * 2;
    unsigned short* wqdb  = (unsigned short*)w;     w += (size_t)Hh * Hh * 2;
    unsigned short* wkdb  = (unsigned short*)w;     w += (size_t)Hh * Hh * 2;
    unsigned short* wvdb  = (unsigned short*)w;     w += (size_t)Hh * Hh * 2;
    unsigned short* wodb  = (unsigned short*)w;     w += (size_t)Hh * Hh * 2;
    unsigned short* dictb = (unsigned short*)w;     w += (size_t)Hh * Ff * 2;

    // output regions: [loss][x_hat][x_novel][z_n]
    float* out  = (float*)d_out;
    float* loss = out;
    float* xhat = out + 1;
    float* xnov = out + 1 + (size_t)NPRED;
    float* zout = out + 1 + (size_t)2 * NPRED;

    zero3<<<1, 32, 0, stream>>>(accum);

    const int wn = Hh * Hh;
    conv_bf16<<<(wn + 255) / 256, 256, 0, stream>>>(Wq_l, wqlb, wn);
    conv_bf16<<<(wn + 255) / 256, 256, 0, stream>>>(Wk_l, wklb, wn);
    conv_bf16<<<(wn + 255) / 256, 256, 0, stream>>>(Wv_l, wvlb, wn);
    conv_bf16<<<(wn + 255) / 256, 256, 0, stream>>>(Wo_l, wolb, wn);
    conv_bf16<<<(wn + 255) / 256, 256, 0, stream>>>(Wq_d, wqdb, wn);
    conv_bf16<<<(wn + 255) / 256, 256, 0, stream>>>(Wk_d, wkdb, wn);
    conv_bf16<<<(wn + 255) / 256, 256, 0, stream>>>(Wv_d, wvdb, wn);
    conv_bf16<<<(wn + 255) / 256, 256, 0, stream>>>(Wo_d, wodb, wn);
    const int dn = Hh * Ff;
    conv_bf16<<<(dn + 255) / 256, 256, 0, stream>>>(dict, dictb, dn);

    hipMemcpyAsync(x_cur, zL, (size_t)M1 * Hh * sizeof(float), hipMemcpyDeviceToDevice, stream);

    dim3 g1(Hh / 64, M1 / 128);   // (12, 128)

    // ---- stage 1: attention over L ----
    layernorm_bf16<<<M1, 256, 0, stream>>>(x_cur, g_l, b_l, lnb);
    gemm_bf16_wmma<<<g1, 256, 0, stream>>>(lnb, wqlb, M1, Hh, Hh, 1, nullptr, 0, nullptr, nullptr, qb);
    gemm_bf16_wmma<<<g1, 256, 0, stream>>>(lnb, wklb, M1, Hh, Hh, 1, nullptr, 0, nullptr, nullptr, kb);
    gemm_bf16_wmma<<<g1, 256, 0, stream>>>(lnb, wvlb, M1, Hh, Hh, 1, nullptr, 0, nullptr, nullptr, vb);
    attn_seq<<<1024, 256, 0, stream>>>(qb, kb, vb, attnb);
    gemm_bf16_wmma<<<g1, 256, 0, stream>>>(attnb, wolb, M1, Hh, Hh, 1, nullptr, 0, x_cur, x_cur, nullptr);

    // ---- stage 2: causal attention over D ----
    const int nel = M1 * Hh;
    transpose_dl<<<nel / 256, 256, 0, stream>>>(x_cur, x2, 0);
    layernorm_bf16<<<M1, 256, 0, stream>>>(x2, g_d, b_d, lnb);
    gemm_bf16_wmma<<<g1, 256, 0, stream>>>(lnb, wqdb, M1, Hh, Hh, 1, nullptr, 0, nullptr, nullptr, qb);
    gemm_bf16_wmma<<<g1, 256, 0, stream>>>(lnb, wkdb, M1, Hh, Hh, 1, nullptr, 0, nullptr, nullptr, kb);
    gemm_bf16_wmma<<<g1, 256, 0, stream>>>(lnb, wvdb, M1, Hh, Hh, 1, nullptr, 0, nullptr, nullptr, vb);
    attn_depth<<<(M1 * 8) / 256, 256, 0, stream>>>(qb, kb, vb, attnb);
    gemm_bf16_wmma<<<g1, 256, 0, stream>>>(attnb, wodb, M1, Hh, Hh, 1, nullptr, 0, x2, x2, nullptr);
    transpose_dl<<<nel / 256, 256, 0, stream>>>(x2, x_cur, 1);

    // ---- sparse autoencoder ----
    residual_pred<<<NPRED / 256, 256, 0, stream>>>(x_cur, resb, accum + 0);
    dim3 g2(Ff / 64, MR / 128);   // (64, 112)
    gemm_bf16_wmma<<<g2, 256, 0, stream>>>(resb, dictb, MR, Ff, Hh, 0, biasn, 1, nullptr, zout, nullptr);
    topk_threshold<<<MR, 256, 0, stream>>>(zout);
    decode_sparse<<<MR, 256, 0, stream>>>(zout, dict, x_cur, xhat, xnov, accum + 1, accum + 2);

    final_loss<<<1, 32, 0, stream>>>(accum, loss);
}
